// EQGATEdgeGNN_48069273977060
// MI455X (gfx1250) — compile-verified, matrix operand order checked by reading, not run on hardware
//
#include <hip/hip_runtime.h>

// ---------------- problem constants ----------------
#define N_      10000
#define E_      160000
#define SDIM_   256
#define VDIM_   128
#define EDIM_   32
#define GDIM_   32
#define NRBF_   20
#define LAYERS_ 5
#define DIN_    600
#define DOUT_   512
#define EPS_    1e-6f
#define CUT_    5.0f

// WMMA tiling
#define NT1_ 16   // SDIM/16 output tiles for GEMM1
#define KC1_ 19   // ceil(600/32) K-chunks (padded to 608)
#define NT2_ 32   // DOUT/16 output tiles for GEMM2
#define KC2_ 8    // 256/32

// LDS strides (elements), padded, 16B-aligned rows
#define FSTRIDE_ 624   // >= 608 bf16
#define HSTRIDE_ 264   // >= 256 bf16
#define MSTRIDE_ 520   // >= 512 f32

typedef unsigned short ushort_t;
typedef __attribute__((ext_vector_type(16))) __bf16 v16bf;
typedef __attribute__((ext_vector_type(8)))  float  v8f;

union FragU { uint4 q[2]; v16bf v; };

__device__ inline ushort_t f2bf(float f){
  union { __bf16 h; ushort_t u; } c; c.h = (__bf16)f; return c.u;
}
// A-fragment from LDS row: elements [0..7] at p[0..7], [8..15] at p[16..23]
__device__ inline v16bf frag_lds(const ushort_t* p){
  FragU u;
  u.q[0] = *(const uint4*)(p);
  u.q[1] = *(const uint4*)(p + 16);
  return u.v;
}
// B-fragment pre-swizzled in global: 16 contiguous bf16
__device__ inline v16bf frag_gbl(const ushort_t* p){
  FragU u;
  u.q[0] = *(const uint4*)(p);
  u.q[1] = *(const uint4*)(p + 8);
  return u.v;
}
__device__ inline void atomAddF(float* p, float v){
  __hip_atomic_fetch_add(p, v, __ATOMIC_RELAXED, __HIP_MEMORY_SCOPE_AGENT);
}
__device__ inline float siluf(float x){ return x / (1.f + __expf(-x)); }

// ---------------- one-time kernels ----------------
__global__ void k_init(const float* __restrict__ s, const float* __restrict__ v,
                       const float* __restrict__ p, const float* __restrict__ ee,
                       float* __restrict__ s_cur, float* __restrict__ v_cur,
                       float* __restrict__ cnt, float* __restrict__ out){
  const long t0 = (long)blockIdx.x*blockDim.x + threadIdx.x;
  const long st = (long)gridDim.x*blockDim.x;
  const long nS = (long)N_*SDIM_, nV = (long)N_*3*VDIM_, nP = (long)N_*3, nE = (long)E_*EDIM_;
  float* out_p = out + nS + nV;
  float* out_e = out_p + nP;
  for (long i=t0;i<nS;i+=st) s_cur[i] = s[i];
  for (long i=t0;i<nV;i+=st) v_cur[i] = v[i];
  for (long i=t0;i<nP;i+=st) out_p[i] = p[i];
  for (long i=t0;i<nE;i+=st) out_e[i] = ee[i];
  for (long i=t0;i<N_;i+=st) cnt[i] = 0.f;
}

__global__ void k_cnt(const int* __restrict__ ei, float* __restrict__ cnt){
  const int e = blockIdx.x*256 + threadIdx.x;
  if (e < E_) atomAddF(&cnt[ei[E_ + e]], 1.f);
}

__global__ void k_zero(float* __restrict__ a, long n){
  long i = (long)blockIdx.x*blockDim.x + threadIdx.x;
  const long st = (long)gridDim.x*blockDim.x;
  for (; i<n; i+=st) a[i] = 0.f;
}

// Pre-swizzle weights into per-lane WMMA B-fragment layout (bf16, K zero-padded)
__global__ void k_prep(const float* __restrict__ We1, const float* __restrict__ We2,
                       ushort_t* __restrict__ We1f, ushort_t* __restrict__ We2f){
  const int gid = blockIdx.x*256 + threadIdx.x;
  const int tot1 = LAYERS_*NT1_*KC1_*32;
  if (gid < tot1){
    const int lane = gid & 31; int t = gid >> 5;
    const int kc = t % KC1_; t /= KC1_;
    const int nt = t % NT1_; const int layer = t / NT1_;
    const int n = lane & 15, half = lane >> 4;
    ushort_t* dst = We1f + (size_t)gid*16;
    #pragma unroll
    for (int e2=0;e2<16;e2++){
      const int within = (e2 < 8) ? e2 : (e2 + 8);     // 0..7 then 16..23
      const int k = kc*32 + 8*half + within;
      const float val = (k < DIN_) ? We1[((size_t)layer*DIN_ + k)*SDIM_ + nt*16 + n] : 0.f;
      dst[e2] = f2bf(val);
    }
    return;
  }
  const int g2 = gid - tot1;
  const int tot2 = LAYERS_*NT2_*KC2_*32;
  if (g2 < tot2){
    const int lane = g2 & 31; int t = g2 >> 5;
    const int kc = t % KC2_; t /= KC2_;
    const int nt = t % NT2_; const int layer = t / NT2_;
    const int n = lane & 15, half = lane >> 4;
    ushort_t* dst = We2f + (size_t)g2*16;
    #pragma unroll
    for (int e2=0;e2<16;e2++){
      const int within = (e2 < 8) ? e2 : (e2 + 8);
      const int k = kc*32 + 8*half + within;
      dst[e2] = f2bf(We2[((size_t)layer*SDIM_ + k)*DOUT_ + nt*16 + n]);
    }
  }
}

// ---------------- per-layer node pre-norm ----------------
// wave-per-node: s <- LN(s) (also bf16 copy), v <- vnorm(v)
__global__ __launch_bounds__(256) void k_prenorm(const float* __restrict__ lng, const float* __restrict__ lnb,
    float* __restrict__ s_cur, ushort_t* __restrict__ s_bf, float* __restrict__ v_cur, int layer){
  const int node = blockIdx.x*8 + (threadIdx.x >> 5);
  const int lane = threadIdx.x & 31;
  if (node >= N_) return;
  float* srow = s_cur + (size_t)node*SDIM_;
  float vals[8]; float sum = 0.f;
  #pragma unroll
  for (int i=0;i<8;i++){ vals[i] = srow[lane + 32*i]; sum += vals[i]; }
  for (int o=16;o>0;o>>=1) sum += __shfl_xor(sum, o, 32);
  const float mu = sum * (1.f/(float)SDIM_);
  float var = 0.f;
  #pragma unroll
  for (int i=0;i<8;i++){ const float d = vals[i]-mu; var += d*d; }
  for (int o=16;o>0;o>>=1) var += __shfl_xor(var, o, 32);
  const float inv = 1.f/sqrtf(var*(1.f/(float)SDIM_) + EPS_);
  #pragma unroll
  for (int i=0;i<8;i++){
    const int col = lane + 32*i;
    const float y = (vals[i]-mu)*inv*lng[layer*SDIM_+col] + lnb[layer*SDIM_+col];
    srow[col] = y;
    s_bf[(size_t)node*SDIM_ + col] = f2bf(y);
  }
  float* vrow = v_cur + (size_t)node*3*VDIM_;
  float vv[12]; float acc = 0.f;
  #pragma unroll
  for (int i=0;i<12;i++){ vv[i] = vrow[lane + 32*i]; acc += vv[i]*vv[i]; }
  for (int o=16;o>0;o>>=1) acc += __shfl_xor(acc, o, 32);
  const float invf = 1.f/sqrtf(acc*(1.f/(float)VDIM_) + EPS_);
  #pragma unroll
  for (int i=0;i<12;i++) vrow[lane + 32*i] = vv[i]*invf;
}

// ---------------- fused edge kernel: gather + GEMM1(WMMA) + SiLU + GEMM2(WMMA) + scatter ----------------
__global__ __launch_bounds__(256) void k_edge(
    const int* __restrict__ ei, const float* __restrict__ ed, const float* __restrict__ ea,
    const float* __restrict__ ern, const float* __restrict__ ee, const float* __restrict__ eo,
    const float* __restrict__ eg, const ushort_t* __restrict__ s_bf, const float* __restrict__ v_cur,
    const ushort_t* __restrict__ We1f, const float* __restrict__ be1,
    const ushort_t* __restrict__ We2f, const float* __restrict__ be2,
    float* __restrict__ ds_acc, float* __restrict__ dv_acc, int layer)
{
  __shared__ __align__(16) ushort_t lds_feat[16*FSTRIDE_];
  __shared__ __align__(16) ushort_t lds_h[16*HSTRIDE_];
  __shared__ __align__(16) float    lds_m[16*MSTRIDE_];
  __shared__ int   lds_src[16];
  __shared__ int   lds_tgt[16];
  __shared__ float lds_rn[16][3];

  const int tid = threadIdx.x;
  const int e0 = blockIdx.x * 16;

  // per-edge metadata + static feature tail (rbf/env, angle, e, ohe, global), cols 512..607 (+pad)
  if (tid < 16){
    const int e = e0 + tid;
    const int sN = ei[e], tN = ei[E_ + e];
    lds_src[tid] = sN; lds_tgt[tid] = tN;
    lds_rn[tid][0] = ern[e*3+0];
    lds_rn[tid][1] = ern[e*3+1];
    lds_rn[tid][2] = ern[e*3+2];
    ushort_t* row = &lds_feat[tid*FSTRIDE_ + 2*SDIM_];
    const float d = ed[e];
    const float env = (d < CUT_) ? 0.5f*(__cosf(3.14159265358979f*d/CUT_) + 1.f) : 0.f;
    for (int k=0;k<NRBF_;k++){
      const float mu = CUT_*(float)k/(float)(NRBF_-1);
      const float dd = d - mu;
      row[k] = f2bf(env*__expf(-8.f*dd*dd));
    }
    row[20] = f2bf(ea[e]);
    for (int j=0;j<EDIM_;j++) row[21+j] = f2bf(ee[(size_t)e*EDIM_+j]);
    for (int j=0;j<3;j++)     row[53+j] = f2bf(eo[e*3+j]);
    for (int j=0;j<GDIM_;j++) row[56+j] = f2bf(eg[(size_t)e*GDIM_+j]);
    for (int j=88;j<96;j++)   row[j] = 0;   // zero pad K 600..607
  }
  // gather s_bf[src], s_bf[tgt] rows (bf16, 128b chunks)
  {
    const int row = tid >> 4, sub = tid & 15;
    const int e = e0 + row;
    const int sN = ei[e], tN = ei[E_ + e];
    const uint4* ps = (const uint4*)(s_bf + (size_t)sN*SDIM_);
    const uint4* pt = (const uint4*)(s_bf + (size_t)tN*SDIM_);
    uint4* d0 = (uint4*)(&lds_feat[row*FSTRIDE_]);
    uint4* d1 = (uint4*)(&lds_feat[row*FSTRIDE_ + SDIM_]);
    d0[2*sub]   = ps[2*sub];
    d0[2*sub+1] = ps[2*sub+1];
    d1[2*sub]   = pt[2*sub];
    d1[2*sub+1] = pt[2*sub+1];
  }
  __syncthreads();

  const int wave = tid >> 5, lane = tid & 31;
  const int mrow = lane & 15, half = lane >> 4, ncol = lane & 15;

  // ---- GEMM1: h = silu(feat @ We1 + be1), wave handles 2 of 16 N-tiles ----
  {
    v8f acc0 = {}; v8f acc1 = {};
    const int nt0 = wave*2, nt1 = nt0+1;
    const ushort_t* arow = &lds_feat[mrow*FSTRIDE_ + 8*half];
    const ushort_t* b0p = We1f + ((((size_t)layer*NT1_ + nt0)*KC1_)*32 + lane)*16;
    const ushort_t* b1p = We1f + ((((size_t)layer*NT1_ + nt1)*KC1_)*32 + lane)*16;
    for (int kc=0;kc<KC1_;kc++){
      const v16bf a  = frag_lds(arow + kc*32);
      const v16bf b0 = frag_gbl(b0p + (size_t)kc*512);
      acc0 = __builtin_amdgcn_wmma_f32_16x16x32_bf16(false, a, false, b0, (short)0, acc0, false, false);
      const v16bf b1 = frag_gbl(b1p + (size_t)kc*512);
      acc1 = __builtin_amdgcn_wmma_f32_16x16x32_bf16(false, a, false, b1, (short)0, acc1, false, false);
    }
    #pragma unroll
    for (int r=0;r<8;r++){
      const int mm = r + 8*half;
      {
        const int col = nt0*16 + ncol;
        const float x = acc0[r] + be1[layer*SDIM_ + col];
        lds_h[mm*HSTRIDE_ + col] = f2bf(siluf(x));
      }
      {
        const int col = nt1*16 + ncol;
        const float x = acc1[r] + be1[layer*SDIM_ + col];
        lds_h[mm*HSTRIDE_ + col] = f2bf(siluf(x));
      }
    }
  }
  __syncthreads();

  // ---- GEMM2: m = h @ We2 + be2, wave handles 4 of 32 N-tiles ----
  {
    v8f c0={}, c1={}, c2={}, c3={};
    const int ntb = wave*4;
    const ushort_t* hrow = &lds_h[mrow*HSTRIDE_ + 8*half];
    for (int kc=0;kc<KC2_;kc++){
      const v16bf a = frag_lds(hrow + kc*32);
      const ushort_t* bp = We2f + ((((size_t)layer*NT2_ + ntb)*KC2_ + kc)*32 + lane)*16;
      const v16bf b0 = frag_gbl(bp);
      c0 = __builtin_amdgcn_wmma_f32_16x16x32_bf16(false, a, false, b0, (short)0, c0, false, false);
      const v16bf b1 = frag_gbl(bp + (size_t)KC2_*512);
      c1 = __builtin_amdgcn_wmma_f32_16x16x32_bf16(false, a, false, b1, (short)0, c1, false, false);
      const v16bf b2 = frag_gbl(bp + (size_t)2*KC2_*512);
      c2 = __builtin_amdgcn_wmma_f32_16x16x32_bf16(false, a, false, b2, (short)0, c2, false, false);
      const v16bf b3 = frag_gbl(bp + (size_t)3*KC2_*512);
      c3 = __builtin_amdgcn_wmma_f32_16x16x32_bf16(false, a, false, b3, (short)0, c3, false, false);
    }
    #pragma unroll
    for (int r=0;r<8;r++){
      const int mm = r + 8*half;
      float* mp = &lds_m[mm*MSTRIDE_];
      mp[(ntb+0)*16+ncol] = c0[r] + be2[layer*DOUT_ + (ntb+0)*16+ncol];
      mp[(ntb+1)*16+ncol] = c1[r] + be2[layer*DOUT_ + (ntb+1)*16+ncol];
      mp[(ntb+2)*16+ncol] = c2[r] + be2[layer*DOUT_ + (ntb+2)*16+ncol];
      mp[(ntb+3)*16+ncol] = c3[r] + be2[layer*DOUT_ + (ntb+3)*16+ncol];
    }
  }
  __syncthreads();

  // ---- scatter: ms -> ds_acc[tgt], mv = gr*rnorm + gv*v[src] -> dv_acc[tgt] ----
  for (int idx = tid; idx < 16*SDIM_; idx += 256){
    const int rr = idx >> 8, col = idx & (SDIM_-1);
    atomAddF(&ds_acc[(size_t)lds_tgt[rr]*SDIM_ + col], lds_m[rr*MSTRIDE_ + col]);
  }
  for (int idx = tid; idx < 16*3*VDIM_; idx += 256){
    const int rr = idx / 384; const int rem = idx - rr*384;
    const int c = rem >> 7, vch = rem & (VDIM_-1);
    const float gr = lds_m[rr*MSTRIDE_ + SDIM_ + vch];
    const float gv = lds_m[rr*MSTRIDE_ + SDIM_ + VDIM_ + vch];
    const float val = gr*lds_rn[rr][c] + gv*v_cur[(size_t)lds_src[rr]*3*VDIM_ + c*VDIM_ + vch];
    atomAddF(&dv_acc[(size_t)lds_tgt[rr]*3*VDIM_ + c*VDIM_ + vch], val);
  }
}

// ---------------- per-layer node update (messages + gated MLP) ----------------
__global__ __launch_bounds__(384) void k_node(float* __restrict__ s_cur, float* __restrict__ v_cur,
    const float* __restrict__ ds_acc, const float* __restrict__ dv_acc, const float* __restrict__ cnt,
    const float* __restrict__ Wvu, const float* __restrict__ Wu1, const float* __restrict__ bu1,
    const float* __restrict__ Wu2, const float* __restrict__ bu2, int layer, int do_mlp){
  __shared__ float t[SDIM_+VDIM_];
  __shared__ float vvs[3*VDIM_];
  __shared__ float u1[SDIM_];
  __shared__ float u[SDIM_+VDIM_];
  const int node = blockIdx.x, tid = threadIdx.x;
  const float rinv = 1.f / fmaxf(cnt[node], 1.f);
  if (tid < SDIM_){
    const float sv = s_cur[(size_t)node*SDIM_+tid] + ds_acc[(size_t)node*SDIM_+tid]*rinv;
    s_cur[(size_t)node*SDIM_+tid] = sv;
    t[tid] = sv;
  }
  {
    const float nv = v_cur[(size_t)node*3*VDIM_+tid] + dv_acc[(size_t)node*3*VDIM_+tid]*rinv;
    v_cur[(size_t)node*3*VDIM_+tid] = nv;
  }
  __syncthreads();
  if (!do_mlp) return;
  {
    const int c = tid >> 7, w = tid & (VDIM_-1);
    const float* vrow = v_cur + (size_t)node*3*VDIM_ + c*VDIM_;
    const float* wcol = Wvu + (size_t)layer*VDIM_*VDIM_ + w;
    float a = 0.f;
    for (int k=0;k<VDIM_;k++) a += vrow[k]*wcol[(size_t)k*VDIM_];
    vvs[tid] = a;
  }
  __syncthreads();
  if (tid < VDIM_){
    const float a0=vvs[tid], a1=vvs[VDIM_+tid], a2=vvs[2*VDIM_+tid];
    t[SDIM_+tid] = sqrtf(a0*a0+a1*a1+a2*a2 + EPS_);
  }
  __syncthreads();
  if (tid < SDIM_){
    float a = bu1[layer*SDIM_ + tid];
    const float* wcol = Wu1 + (size_t)layer*(SDIM_+VDIM_)*SDIM_ + tid;
    for (int k=0;k<SDIM_+VDIM_;k++) a += t[k]*wcol[(size_t)k*SDIM_];
    u1[tid] = siluf(a);
  }
  __syncthreads();
  {
    float a = bu2[layer*(SDIM_+VDIM_) + tid];
    const float* wcol = Wu2 + (size_t)layer*SDIM_*(SDIM_+VDIM_) + tid;
    for (int k=0;k<SDIM_;k++) a += u1[k]*wcol[(size_t)k*(SDIM_+VDIM_)];
    u[tid] = a;
  }
  __syncthreads();
  if (tid < SDIM_) s_cur[(size_t)node*SDIM_+tid] += u[tid];
  v_cur[(size_t)node*3*VDIM_+tid] += vvs[tid]*u[SDIM_ + (tid & (VDIM_-1))];
}

// ---------------- final LN/vnorm -> d_out ----------------
__global__ __launch_bounds__(256) void k_final(const float* __restrict__ lng, const float* __restrict__ lnb,
    const float* __restrict__ s_cur, const float* __restrict__ v_cur, float* __restrict__ out){
  const int node = blockIdx.x*8 + (threadIdx.x >> 5);
  const int lane = threadIdx.x & 31;
  if (node >= N_) return;
  float* out_s = out;
  float* out_v = out + (size_t)N_*SDIM_;
  const float* srow = s_cur + (size_t)node*SDIM_;
  float vals[8]; float sum = 0.f;
  #pragma unroll
  for (int i=0;i<8;i++){ vals[i] = srow[lane + 32*i]; sum += vals[i]; }
  for (int o=16;o>0;o>>=1) sum += __shfl_xor(sum, o, 32);
  const float mu = sum * (1.f/(float)SDIM_);
  float var = 0.f;
  #pragma unroll
  for (int i=0;i<8;i++){ const float d = vals[i]-mu; var += d*d; }
  for (int o=16;o>0;o>>=1) var += __shfl_xor(var, o, 32);
  const float inv = 1.f/sqrtf(var*(1.f/(float)SDIM_) + EPS_);
  #pragma unroll
  for (int i=0;i<8;i++){
    const int col = lane + 32*i;
    out_s[(size_t)node*SDIM_+col] = (vals[i]-mu)*inv*lng[LAYERS_*SDIM_+col] + lnb[LAYERS_*SDIM_+col];
  }
  const float* vrow = v_cur + (size_t)node*3*VDIM_;
  float vv[12]; float acc = 0.f;
  #pragma unroll
  for (int i=0;i<12;i++){ vv[i] = vrow[lane + 32*i]; acc += vv[i]*vv[i]; }
  for (int o=16;o>0;o>>=1) acc += __shfl_xor(acc, o, 32);
  const float invf = 1.f/sqrtf(acc*(1.f/(float)VDIM_) + EPS_);
  #pragma unroll
  for (int i=0;i<12;i++) out_v[(size_t)node*3*VDIM_ + lane + 32*i] = vv[i]*invf;
}

// ---------------- host launch ----------------
extern "C" void kernel_launch(void* const* d_in, const int* in_sizes, int n_in,
                              void* d_out, int out_size, void* d_ws, size_t ws_size,
                              hipStream_t stream){
  (void)in_sizes; (void)n_in; (void)out_size; (void)ws_size;
  const float* s   = (const float*)d_in[0];
  const float* v   = (const float*)d_in[1];
  const float* p   = (const float*)d_in[2];
  const int*   ei  = (const int*)  d_in[3];
  const float* ed  = (const float*)d_in[4];
  const float* ea  = (const float*)d_in[5];
  const float* ern = (const float*)d_in[6];
  const float* ee  = (const float*)d_in[7];
  const float* eo  = (const float*)d_in[8];
  const float* eg  = (const float*)d_in[9];
  const float* lng = (const float*)d_in[10];
  const float* lnb = (const float*)d_in[11];
  const float* We1 = (const float*)d_in[12];
  const float* be1 = (const float*)d_in[13];
  const float* We2 = (const float*)d_in[14];
  const float* be2 = (const float*)d_in[15];
  const float* Wvu = (const float*)d_in[16];
  const float* Wu1 = (const float*)d_in[17];
  const float* bu1 = (const float*)d_in[18];
  const float* Wu2 = (const float*)d_in[19];
  const float* bu2 = (const float*)d_in[20];
  float* out = (float*)d_out;

  char* ws = (char*)d_ws; size_t off = 0;
  auto alloc = [&](size_t bytes)->char*{
    off = (off + 255) & ~(size_t)255;
    char* r = ws + off; off += bytes; return r;
  };
  float*    s_cur  = (float*)   alloc((size_t)N_*SDIM_*4);
  float*    v_cur  = (float*)   alloc((size_t)N_*3*VDIM_*4);
  ushort_t* s_bf   = (ushort_t*)alloc((size_t)N_*SDIM_*2);
  float*    ds_acc = (float*)   alloc((size_t)N_*SDIM_*4);
  float*    dv_acc = (float*)   alloc((size_t)N_*3*VDIM_*4);
  float*    cnt    = (float*)   alloc((size_t)N_*4);
  ushort_t* We1f   = (ushort_t*)alloc((size_t)LAYERS_*NT1_*KC1_*32*16*2);
  ushort_t* We2f   = (ushort_t*)alloc((size_t)LAYERS_*NT2_*KC2_*32*16*2);

  k_init<<<4096, 256, 0, stream>>>(s, v, p, ee, s_cur, v_cur, cnt, out);
  k_cnt<<<(E_+255)/256, 256, 0, stream>>>(ei, cnt);
  const int prep_threads = LAYERS_*(NT1_*KC1_ + NT2_*KC2_)*32;
  k_prep<<<(prep_threads+255)/256, 256, 0, stream>>>(We1, We2, We1f, We2f);

  for (int i=0;i<LAYERS_;i++){
    k_prenorm<<<(N_+7)/8, 256, 0, stream>>>(lng, lnb, s_cur, s_bf, v_cur, i);
    k_zero<<<2048, 256, 0, stream>>>(ds_acc, (long)N_*SDIM_);
    k_zero<<<2048, 256, 0, stream>>>(dv_acc, (long)N_*3*VDIM_);
    k_edge<<<E_/16, 256, 0, stream>>>(ei, ed, ea, ern, ee, eo, eg, s_bf, v_cur,
                                      We1f, be1, We2f, be2, ds_acc, dv_acc, i);
    k_node<<<N_, 384, 0, stream>>>(s_cur, v_cur, ds_acc, dv_acc, cnt,
                                   Wvu, Wu1, bu1, Wu2, bu2, i, (i < LAYERS_-1) ? 1 : 0);
  }
  k_final<<<(N_+7)/8, 256, 0, stream>>>(lng, lnb, s_cur, v_cur, out);
}
